// GPT2_78993038508852
// MI455X (gfx1250) — matrix operand order repaired and checked by reading, not compiled, not run on hardware
//
#include <hip/hip_runtime.h>
#include <hip/hip_bf16.h>

// ---------------------------------------------------------------------------
// Types for CDNA5 WMMA (wave32): V_WMMA_F32_16X16X32_BF16
// ---------------------------------------------------------------------------
typedef __attribute__((ext_vector_type(16))) __bf16 v16bf;
typedef __attribute__((ext_vector_type(8)))  float  v8f;

union U16x16 {
  unsigned short us[16];
  uint4          u4[2];
  v16bf          v;
};

__device__ __forceinline__ unsigned short f2bf(float f) {
  unsigned int u = __float_as_uint(f);
  u += 0x7FFFu + ((u >> 16) & 1u);   // round-to-nearest-even
  return (unsigned short)(u >> 16);
}

__device__ __forceinline__ v8f wmma_bf16(v16bf a, v16bf b, v8f c) {
  // 8 args: (neg_a, A, neg_b, B, c_mod, C, reuse_a, reuse_b)
  return __builtin_amdgcn_wmma_f32_16x16x32_bf16(false, a, false, b, (short)0, c,
                                                 false, false);
}

__device__ __forceinline__ float gelu_tanh(float x) {
  float x3 = x * x * x;
  return 0.5f * x * (1.0f + tanhf(0.7978845608028654f * (x + 0.044715f * x3)));
}

// Async copy 16 bytes global -> LDS (GLOBAL_LOAD_ASYNC_TO_LDS_B128, ASYNCcnt).
// LDS operand is the 32-bit LDS offset = low 32 bits of the generic pointer.
__device__ __forceinline__ void async_copy_b128(const unsigned short* gsrc,
                                                unsigned short* ldst) {
  unsigned lds = (unsigned)(size_t)ldst;
  unsigned long long ga = (unsigned long long)(size_t)gsrc;
  asm volatile("global_load_async_to_lds_b128 %0, %1, off"
               :: "v"(lds), "v"(ga) : "memory");
}

__device__ __forceinline__ void wait_async_all() {
  asm volatile("s_wait_asynccnt 0x0" ::: "memory");
}

#define GF_BTRANS 1
#define GF_GELU   2

// ---------------------------------------------------------------------------
// Weight cast fp32 -> bf16
// ---------------------------------------------------------------------------
__global__ void cast_f32_bf16(const float* __restrict__ in,
                              unsigned short* __restrict__ out, size_t n) {
  size_t i = blockIdx.x * (size_t)blockDim.x + threadIdx.x;
  size_t stride = gridDim.x * (size_t)blockDim.x;
  for (; i < n; i += stride) out[i] = f2bf(in[i]);
}

// ---------------------------------------------------------------------------
// Token + position embedding
// ---------------------------------------------------------------------------
__global__ void embed_add(const int* __restrict__ ids, const float* __restrict__ tok,
                          const float* __restrict__ pos, float* __restrict__ hout,
                          int Cn, int Tn, size_t total) {
  size_t i = blockIdx.x * (size_t)blockDim.x + threadIdx.x;
  if (i >= total) return;
  size_t bt = i / Cn;
  int cc = (int)(i % Cn);
  int tt = (int)(bt % Tn);
  hout[i] = tok[(size_t)ids[bt] * Cn + cc] + pos[(size_t)tt * Cn + cc];
}

// ---------------------------------------------------------------------------
// LayerNorm (fp32 in -> bf16 out), one row (C=768) per 256-thread block
// ---------------------------------------------------------------------------
__global__ __launch_bounds__(256) void layernorm_bf16(
    const float* __restrict__ x, const float* __restrict__ w,
    const float* __restrict__ bparm, unsigned short* __restrict__ out, int Cn) {
  __shared__ float red[256];
  const int row = blockIdx.x;
  const int tid = threadIdx.x;
  const float* xr = x + (size_t)row * Cn;

  float s = 0.0f;
  for (int i = tid; i < Cn; i += 256) s += xr[i];
  red[tid] = s; __syncthreads();
  for (int off = 128; off > 0; off >>= 1) {
    if (tid < off) red[tid] += red[tid + off];
    __syncthreads();
  }
  const float mean = red[0] / Cn;
  __syncthreads();

  float vs = 0.0f;
  for (int i = tid; i < Cn; i += 256) { float d = xr[i] - mean; vs += d * d; }
  red[tid] = vs; __syncthreads();
  for (int off = 128; off > 0; off >>= 1) {
    if (tid < off) red[tid] += red[tid + off];
    __syncthreads();
  }
  const float rstd = rsqrtf(red[0] / Cn + 1e-5f);

  for (int i = tid; i < Cn; i += 256)
    out[(size_t)row * Cn + i] = f2bf((xr[i] - mean) * rstd * w[i] + bparm[i]);
}

// ---------------------------------------------------------------------------
// bf16 WMMA GEMM: out[M,N] = A[M,K] * B[K,N] (+bias)(+residual)(+gelu)
// 256 threads = 8 waves; block tile 128x128; wave tile 32x64 (2x4 WMMA tiles)
// K stepped by 32 through LDS. A tile (and B tile in BTRANS mode) staged with
// GLOBAL_LOAD_ASYNC_TO_LDS_B128 (no VGPR round-trip, overlaps prior WMMAs).
// flags: GF_BTRANS (B stored [N,K]), GF_GELU.
// ---------------------------------------------------------------------------
__global__ __launch_bounds__(256) void gemm_bf16_wmma(
    const unsigned short* __restrict__ A, const unsigned short* __restrict__ B,
    const float* __restrict__ bias, const float* __restrict__ residual,
    float* __restrict__ outF, unsigned short* __restrict__ outH,
    int M, int N, int K, int lda, int ldb, int flags) {
  __shared__ unsigned short As[128 * 40];   // [row][k], stride 40 halfs (80B)
  __shared__ unsigned short Bs[128 * 40];   // [col][k], transposed in LDS

  const int tid    = threadIdx.x;
  const int lane   = tid & 31;
  const int wave   = tid >> 5;
  const int wm     = wave & 3;        // wave M coord (4)
  const int wn     = wave >> 2;       // wave N coord (2)
  const int l15    = lane & 15;
  const int hi     = lane >> 4;       // lane half-group
  const int khalfA = hi * 8;          // A frag: runs {khalf..+7, khalf+16..+23}
  const int kbB    = hi * 8;          // B frag: runs {kb..+7, kb+16..+23}

  const int n0 = blockIdx.x * 128;
  const int m0 = blockIdx.y * 128;

  v8f acc[2][4] = {{{}, {}, {}, {}}, {{}, {}, {}, {}}};

  const int ksteps = K >> 5;
  for (int ks = 0; ks < ksteps; ++ks) {
    const int k0 = ks << 5;

    // --- stage A tile 128x32 (row-major) via async-to-LDS ---
#pragma unroll
    for (int j = 0; j < 2; ++j) {
      int c = j * 256 + tid;            // 512 chunks of 8 halfs
      int row = c >> 2, kg = c & 3;
      const unsigned short* gsrc = A + (size_t)(m0 + row) * lda + k0 + kg * 8;
      async_copy_b128(gsrc, &As[row * 40 + kg * 8]);
      if (ks + 1 < ksteps) __builtin_prefetch(gsrc + 32, 0, 1);
    }

    // --- stage B tile 32x128, transposed into LDS as [n][k] ---
    if (flags & GF_BTRANS) {
      // B stored [N,K] row-major (tok_embed for tied LM head): contiguous in k
#pragma unroll
      for (int j = 0; j < 2; ++j) {
        int c = j * 256 + tid;
        int nn = c >> 2, kg = c & 3;
        int gn = n0 + nn;
        if (gn < N)
          async_copy_b128(B + (size_t)gn * ldb + k0 + kg * 8,
                          &Bs[nn * 40 + kg * 8]);
      }
    } else {
      // B stored [K,N] row-major; coalesced over n, scatter-transpose into LDS
#pragma unroll
      for (int i = 0; i < 16; ++i) {
        int c = i * 256 + tid;          // 4096 elements
        int kk = c >> 7, nn = c & 127;
        int gn = n0 + nn;
        unsigned short vv =
            (gn < N) ? B[(size_t)(k0 + kk) * ldb + gn] : (unsigned short)0;
        Bs[nn * 40 + kk] = vv;
      }
    }
    wait_async_all();      // our asm-issued asyncs aren't known to the fence
    __syncthreads();

    // --- fragments + 8 WMMAs ---
    U16x16 af[2], bfrag[4];
#pragma unroll
    for (int mi = 0; mi < 2; ++mi) {
      int r = wm * 32 + mi * 16 + l15;
      af[mi].u4[0] = *reinterpret_cast<const uint4*>(&As[r * 40 + khalfA]);
      af[mi].u4[1] = *reinterpret_cast<const uint4*>(&As[r * 40 + khalfA + 16]);
    }
#pragma unroll
    for (int ni = 0; ni < 4; ++ni) {
      int cc = wn * 64 + ni * 16 + l15;
      bfrag[ni].u4[0] = *reinterpret_cast<const uint4*>(&Bs[cc * 40 + kbB]);
      bfrag[ni].u4[1] = *reinterpret_cast<const uint4*>(&Bs[cc * 40 + kbB + 16]);
    }
#pragma unroll
    for (int mi = 0; mi < 2; ++mi)
#pragma unroll
      for (int ni = 0; ni < 4; ++ni)
        acc[mi][ni] = wmma_bf16(af[mi].v, bfrag[ni].v, acc[mi][ni]);
    __syncthreads();
  }

  // --- epilogue: C/D layout rows m = i + 8*hi, col = l15 within tile ---
#pragma unroll
  for (int mi = 0; mi < 2; ++mi) {
#pragma unroll
    for (int ni = 0; ni < 4; ++ni) {
      int colg = n0 + wn * 64 + ni * 16 + l15;
      if (colg >= N) continue;
      int rowb = m0 + wm * 32 + mi * 16 + hi * 8;
      float bv = bias ? bias[colg] : 0.0f;
#pragma unroll
      for (int i = 0; i < 8; ++i) {
        int r = rowb + i;
        float v = acc[mi][ni][i] + bv;
        if (residual) v += residual[(size_t)r * N + colg];
        if (flags & GF_GELU) v = gelu_tanh(v);
        if (outH) outH[(size_t)r * N + colg] = f2bf(v);
        else      outF[(size_t)r * N + colg] = v;
      }
    }
  }
}

// ---------------------------------------------------------------------------
// Flash attention: one wave per (b, h, 16-row q tile). Causal, online softmax.
// qkv layout: [B*T, 3C] bf16; Q at +0, K at +C, V at +2C within each row.
// ---------------------------------------------------------------------------
__global__ __launch_bounds__(32) void attn_flash(
    const unsigned short* __restrict__ qkv, unsigned short* __restrict__ out) {
  const int qt = blockIdx.x, hh = blockIdx.y, bb = blockIdx.z;
  const int lane = threadIdx.x;
  const int l15 = lane & 15, hi = lane >> 4;
  const int khalf = hi * 8;    // A-frag K offsets
  const int kbB   = hi * 8;    // B-frag K base
  __shared__ unsigned short ps[16 * 40];   // P tile 16x32 bf16 (stride 40)

  const int Tn = 1024, C3 = 2304, Cc = 768, DH = 64;
  const size_t baseRow = (size_t)bb * Tn;

  // Q A-fragments (16x64 = two k=32 fragments), contiguous per-lane loads
  U16x16 qa[2];
  {
    const unsigned short* qrow =
        qkv + (baseRow + qt * 16 + l15) * C3 + (size_t)hh * DH;
#pragma unroll
    for (int f = 0; f < 2; ++f) {
      qa[f].u4[0] = *reinterpret_cast<const uint4*>(qrow + f * 32 + khalf);
      qa[f].u4[1] = *reinterpret_cast<const uint4*>(qrow + f * 32 + khalf + 16);
    }
  }

  v8f o[4] = {{}, {}, {}, {}};
  float mrun[8], lrun[8];
#pragma unroll
  for (int i = 0; i < 8; ++i) { mrun[i] = -3.0e38f; lrun[i] = 0.0f; }

  const int q0 = qt * 16;
  const int ntEnd = (q0 + 15) >> 5;
  for (int nt = 0; nt <= ntEnd; ++nt) {
    const int kv0 = nt * 32;

    // S = Q @ K^T : two 16x16 C-tiles (cols kv0..kv0+15, kv0+16..kv0+31)
    v8f s[2];
#pragma unroll
    for (int nh = 0; nh < 2; ++nh) {
      const unsigned short* krow =
          qkv + (baseRow + kv0 + nh * 16 + l15) * C3 + Cc + (size_t)hh * DH;
      v8f z = {};
#pragma unroll
      for (int f = 0; f < 2; ++f) {
        U16x16 kb;
        kb.u4[0] = *reinterpret_cast<const uint4*>(krow + f * 32 + kbB);
        kb.u4[1] = *reinterpret_cast<const uint4*>(krow + f * 32 + kbB + 16);
        z = wmma_bf16(qa[f].v, kb.v, z);
      }
      s[nh] = z;
    }

    // scale, causal mask, online softmax update (rows span 16 lanes)
    const float sc = 0.125f;   // 1/sqrt(64)
#pragma unroll
    for (int i = 0; i < 8; ++i) {
      const int row = q0 + hi * 8 + i;
#pragma unroll
      for (int nh = 0; nh < 2; ++nh) {
        const int col = kv0 + nh * 16 + l15;
        float v = s[nh][i] * sc;
        s[nh][i] = (col <= row) ? v : -3.0e38f;
      }
      float mx = fmaxf(s[0][i], s[1][i]);
#pragma unroll
      for (int d = 1; d < 16; d <<= 1) mx = fmaxf(mx, __shfl_xor(mx, d, 16));
      const float mnew = fmaxf(mrun[i], mx);
      const float p0 = __expf(s[0][i] - mnew);
      const float p1 = __expf(s[1][i] - mnew);
      float rs = p0 + p1;
#pragma unroll
      for (int d = 1; d < 16; d <<= 1) rs += __shfl_xor(rs, d, 16);
      const float alpha = __expf(mrun[i] - mnew);
      lrun[i] = lrun[i] * alpha + rs;
      mrun[i] = mnew;
#pragma unroll
      for (int ni = 0; ni < 4; ++ni) o[ni][i] *= alpha;
      s[0][i] = p0; s[1][i] = p1;
    }

    // P (C-layout) -> LDS -> A-layout fragment
#pragma unroll
    for (int nh = 0; nh < 2; ++nh)
#pragma unroll
      for (int i = 0; i < 8; ++i)
        ps[(hi * 8 + i) * 40 + nh * 16 + l15] = f2bf(s[nh][i]);
    __syncthreads();
    U16x16 pa;
    pa.u4[0] = *reinterpret_cast<const uint4*>(&ps[l15 * 40 + khalf]);
    pa.u4[1] = *reinterpret_cast<const uint4*>(&ps[l15 * 40 + khalf + 16]);

    // O += P @ V : V B-fragments gathered per-lane (k strided by 3C)
#pragma unroll
    for (int ni = 0; ni < 4; ++ni) {
      U16x16 vb;
#pragma unroll
      for (int j = 0; j < 16; ++j) {
        int kk = kbB + (j & 7) + ((j >> 3) << 4);
        vb.us[j] = qkv[(baseRow + kv0 + kk) * C3 + 2 * Cc + (size_t)hh * DH +
                       ni * 16 + l15];
      }
      o[ni] = wmma_bf16(pa.v, vb.v, o[ni]);
    }
    __syncthreads();
  }

  // write O / l  -> attn_out [B*T, C] bf16
#pragma unroll
  for (int ni = 0; ni < 4; ++ni)
#pragma unroll
    for (int i = 0; i < 8; ++i) {
      int t = q0 + hi * 8 + i;
      out[(baseRow + t) * Cc + (size_t)hh * DH + ni * 16 + l15] =
          f2bf(o[ni][i] / lrun[i]);
    }
}

// ---------------------------------------------------------------------------
// Host orchestration
// ---------------------------------------------------------------------------
extern "C" void kernel_launch(void* const* d_in, const int* in_sizes, int n_in,
                              void* d_out, int out_size, void* d_ws, size_t ws_size,
                              hipStream_t stream) {
  constexpr int Vv = 50257, Cc = 768, FFf = 3072, Hh = 12, Ll = 12, Bb = 4, Tt = 1024;
  constexpr int Mm = Bb * Tt;   // 4096

  const int*   x_ids = (const int*)d_in[0];
  const float* tok   = (const float*)d_in[1];
  const float* pos   = (const float*)d_in[2];
  const float* ln1w  = (const float*)d_in[3];
  const float* ln1b  = (const float*)d_in[4];
  const float* qkvw  = (const float*)d_in[5];
  const float* qkvb  = (const float*)d_in[6];
  const float* projw = (const float*)d_in[7];
  const float* projb = (const float*)d_in[8];
  const float* ln2w  = (const float*)d_in[9];
  const float* ln2b  = (const float*)d_in[10];
  const float* fcw   = (const float*)d_in[11];
  const float* fcb   = (const float*)d_in[12];
  const float* fcpw  = (const float*)d_in[13];
  const float* fcpb  = (const float*)d_in[14];
  const float* lnfw  = (const float*)d_in[15];
  const float* lnfb  = (const float*)d_in[16];

  char* p = (char*)d_ws;
  auto take = [&](size_t bytes) {
    char* r = p;
    p += (bytes + 255) & ~(size_t)255;
    return r;
  };

  unsigned short* wTok  = (unsigned short*)take((size_t)Vv * Cc * 2);
  unsigned short* wQkv  = (unsigned short*)take((size_t)Ll * Cc * 3 * Cc * 2);
  unsigned short* wProj = (unsigned short*)take((size_t)Ll * Cc * Cc * 2);
  unsigned short* wFc   = (unsigned short*)take((size_t)Ll * Cc * FFf * 2);
  unsigned short* wFcp  = (unsigned short*)take((size_t)Ll * FFf * Cc * 2);
  float*          hbuf  = (float*)take((size_t)Mm * Cc * 4);
  unsigned short* xn    = (unsigned short*)take((size_t)Mm * Cc * 2);
  unsigned short* qkvB  = (unsigned short*)take((size_t)Mm * 3 * Cc * 2);
  unsigned short* attnB = (unsigned short*)take((size_t)Mm * Cc * 2);
  unsigned short* ffB   = (unsigned short*)take((size_t)Mm * FFf * 2);

  // 1) weights -> bf16 (deterministic, every call)
  cast_f32_bf16<<<2048, 256, 0, stream>>>(tok, wTok, (size_t)Vv * Cc);
  cast_f32_bf16<<<2048, 256, 0, stream>>>(qkvw, wQkv, (size_t)Ll * Cc * 3 * Cc);
  cast_f32_bf16<<<2048, 256, 0, stream>>>(projw, wProj, (size_t)Ll * Cc * Cc);
  cast_f32_bf16<<<2048, 256, 0, stream>>>(fcw, wFc, (size_t)Ll * Cc * FFf);
  cast_f32_bf16<<<2048, 256, 0, stream>>>(fcpw, wFcp, (size_t)Ll * FFf * Cc);

  // 2) embeddings
  {
    size_t total = (size_t)Mm * Cc;
    embed_add<<<(unsigned)((total + 255) / 256), 256, 0, stream>>>(
        x_ids, tok, pos, hbuf, Cc, Tt, total);
  }

  // 3) transformer blocks
  for (int l = 0; l < Ll; ++l) {
    layernorm_bf16<<<Mm, 256, 0, stream>>>(hbuf, ln1w + (size_t)l * Cc,
                                           ln1b + (size_t)l * Cc, xn, Cc);
    {  // qkv = xn @ W_qkv + b  -> bf16
      dim3 g((3 * Cc + 127) / 128, Mm / 128);
      gemm_bf16_wmma<<<g, 256, 0, stream>>>(
          xn, wQkv + (size_t)l * Cc * 3 * Cc, qkvb + (size_t)l * 3 * Cc,
          nullptr, nullptr, qkvB, Mm, 3 * Cc, Cc, Cc, 3 * Cc, 0);
    }
    attn_flash<<<dim3(Tt / 16, Hh, Bb), 32, 0, stream>>>(qkvB, attnB);
    {  // h = h + attn @ W_proj + b
      dim3 g((Cc + 127) / 128, Mm / 128);
      gemm_bf16_wmma<<<g, 256, 0, stream>>>(
          attnB, wProj + (size_t)l * Cc * Cc, projb + (size_t)l * Cc,
          hbuf, hbuf, nullptr, Mm, Cc, Cc, Cc, Cc, 0);
    }
    layernorm_bf16<<<Mm, 256, 0, stream>>>(hbuf, ln2w + (size_t)l * Cc,
                                           ln2b + (size_t)l * Cc, xn, Cc);
    {  // ff = gelu(xn @ W_fc + b) -> bf16
      dim3 g((FFf + 127) / 128, Mm / 128);
      gemm_bf16_wmma<<<g, 256, 0, stream>>>(
          xn, wFc + (size_t)l * Cc * FFf, fcb + (size_t)l * FFf,
          nullptr, nullptr, ffB, Mm, FFf, Cc, Cc, FFf, GF_GELU);
    }
    {  // h = h + ff @ W_fcp + b
      dim3 g((Cc + 127) / 128, Mm / 128);
      gemm_bf16_wmma<<<g, 256, 0, stream>>>(
          ffB, wFcp + (size_t)l * FFf * Cc, fcpb + (size_t)l * Cc,
          hbuf, hbuf, nullptr, Mm, Cc, FFf, FFf, Cc, 0);
    }
  }

  // 4) final LN + tied LM head (B = tok_embed^T, bounds-guarded N=50257)
  layernorm_bf16<<<Mm, 256, 0, stream>>>(hbuf, lnfw, lnfb, xn, Cc);
  {
    dim3 g((Vv + 127) / 128, Mm / 128);
    gemm_bf16_wmma<<<g, 256, 0, stream>>>(
        xn, wTok, nullptr, nullptr, (float*)d_out, nullptr,
        Mm, Vv, Cc, Cc, Cc, GF_BTRANS);
  }

  (void)in_sizes; (void)n_in; (void)out_size; (void)ws_size;
}
